// self_attention_with_bias_6571299963149
// MI455X (gfx1250) — compile-verified, matrix operand order checked by reading, not run on hardware
//
#include <hip/hip_runtime.h>
#include <hip/hip_bf16.h>
#include <math.h>

// ---------------- types ----------------
typedef _Float16 half_t;
typedef __attribute__((ext_vector_type(8)))  _Float16 v8h;
typedef __attribute__((ext_vector_type(16))) _Float16 v16h;
typedef __attribute__((ext_vector_type(8)))  float    v8f;

#define D_MODEL 1024
#define N_HEAD  16
#define HEAD_DIM 64
#define BS 2
#define SEQ 2048
#define NTOK (BS*SEQ)            // 4096
#define BH   (BS*N_HEAD)         // 32
#define LSTRIDE (SEQ + 4)        // logits row stride: 2052 ≡ 4 mod 64 banks

static __device__ inline v16h cat16(v8h lo, v8h hi) {
    return __builtin_shufflevector(lo, hi, 0,1,2,3,4,5,6,7,8,9,10,11,12,13,14,15);
}

static __device__ inline v8f wmma_f16(v16h a, v16h b, v8f c) {
    // (neg_a, A, neg_b, B, c_mod, C, reuse_a, reuse_b)
    return __builtin_amdgcn_wmma_f32_16x16x32_f16(false, a, false, b, (short)0, c, false, false);
}

// ---------------- stage 1: f32 -> f16 convert ----------------
__global__ void f32_to_f16_kernel(const float* __restrict__ in, half_t* __restrict__ out, int n) {
    int i = blockIdx.x * blockDim.x + threadIdx.x;
    if (i < n) out[i] = (half_t)in[i];
}

// ---------------- stage 2: QKV projection GEMM ----------------
// y = x @ W^T + b.  One wave per 16x16 output tile, K-loop of 32.
// A-operand: lane = row (M), halves hold K runs {0..7,16..23} / {8..15,24..31}.
// B-operand (transpose-symmetric): lane = output column (N) = W row, same K gather.
__global__ __launch_bounds__(256)
void qkv_gemm_kernel(const half_t* __restrict__ xh,
                     const half_t* __restrict__ wqh, const half_t* __restrict__ wkh,
                     const half_t* __restrict__ wvh,
                     const float* __restrict__ bq, const float* __restrict__ bk,
                     const float* __restrict__ bv,
                     half_t* __restrict__ qh, half_t* __restrict__ kh,
                     half_t* __restrict__ vth) {
    const int which = blockIdx.y;                 // 0=Q 1=K 2=V
    const half_t* wh = (which == 0) ? wqh : (which == 1) ? wkh : wvh;
    const float*  bb = (which == 0) ? bq  : (which == 1) ? bk  : bv;

    const int wave = threadIdx.x >> 5;
    const int lane = threadIdx.x & 31;
    const int tile = blockIdx.x * 8 + wave;       // 0..16383
    const int m0 = (tile >> 6) * 16;              // 256 m-tiles
    const int n0 = (tile & 63) * 16;              // 64 n-tiles

    const int row  = lane & 15;
    const int koff = (lane >> 4) * 8;
    const half_t* aptr = xh + (size_t)(m0 + row) * D_MODEL;
    const half_t* bptr = wh + (size_t)(n0 + row) * D_MODEL;

    v8f c = {};
    for (int ks = 0; ks < D_MODEL; ks += 32) {
        const int kb = ks + koff;
        v16h a = cat16(*(const v8h*)(aptr + kb), *(const v8h*)(aptr + kb + 16));
        v16h b = cat16(*(const v8h*)(bptr + kb), *(const v8h*)(bptr + kb + 16));
        c = wmma_f16(a, b, c);
    }

    const int ncol = n0 + (lane & 15);            // output feature j
    const float bval = bb[ncol];
    const int h = ncol >> 6, d = ncol & 63;
#pragma unroll
    for (int r = 0; r < 8; ++r) {
        const int i = m0 + r + ((lane >> 4) << 3);   // token index
        const int bidx = i >> 11, s = i & (SEQ - 1);
        const half_t val = (half_t)(c[r] + bval);
        if (which == 2)
            vth[((size_t)(bidx * N_HEAD + h) * HEAD_DIM + d) * SEQ + s] = val;  // V^T: [bh][d][s]
        else {
            half_t* dst = (which == 0) ? qh : kh;
            dst[((size_t)(bidx * N_HEAD + h) * SEQ + s) * HEAD_DIM + d] = val;  // [bh][s][d]
        }
    }
}

// ---------------- stage 3: attention ----------------
// Block = 4 waves = 128 threads. One block per (b,h, 16-query tile).
// LDS: 16x2052 fp32 logits (padded stride for banking) + 256 f32 scratch ~= 129.3 KB.
__global__ __launch_bounds__(128)
void attn_kernel(const half_t* __restrict__ qh, const half_t* __restrict__ kh,
                 const half_t* __restrict__ vth,
                 const float* __restrict__ bias, const int* __restrict__ mask,
                 float* __restrict__ out, float* __restrict__ attnw) {
    extern __shared__ float lds[];
    float* logits  = lds;                  // 16 * LSTRIDE
    float* scratch = lds + 16 * LSTRIDE;   // 256

    const int qt = blockIdx.x;             // 0..127 query tile
    const int bh = blockIdx.y;             // 0..31
    const int h  = bh & (N_HEAD - 1);
    const int tid = threadIdx.x, wave = tid >> 5, lane = tid & 31;
    const int q0 = qt * 16;

    const half_t* Q  = qh  + (size_t)bh * SEQ * HEAD_DIM;
    const half_t* K  = kh  + (size_t)bh * SEQ * HEAD_DIM;
    const half_t* VT = vth + (size_t)bh * HEAD_DIM * SEQ;

    const int row  = lane & 15;
    const int koff = (lane >> 4) * 8;

    // preload Q tile A-operands (d = 0..31 and 32..63), reused for all 128 key tiles
    const half_t* qrow = Q + (size_t)(q0 + row) * HEAD_DIM;
    const v16h a0 = cat16(*(const v8h*)(qrow + koff),      *(const v8h*)(qrow + koff + 16));
    const v16h a1 = cat16(*(const v8h*)(qrow + 32 + koff), *(const v8h*)(qrow + 32 + koff + 16));

    const float scale = 0.125f;   // 1/sqrt(64)

    // ---- phase 1: logits = QK^T * scale + bias, masked, into LDS ----
    for (int nt = wave; nt < SEQ / 16; nt += 4) {
        const int n0 = nt * 16;

        // prefetch next key-tile's bias/mask lines (this wave's nt+4 -> n0+64):
        // lane&15 spreads over the 16 query rows, lane>>4 covers both 32B halves of the 64B span
        const int pk = n0 + 64;
        if (pk < SEQ) {
            const int prw = q0 + (lane & 15);
            const int pcl = pk + ((lane >> 4) << 3);
            __builtin_prefetch(&bias[((size_t)h * SEQ + prw) * SEQ + pcl], 0, 1);
            __builtin_prefetch(&mask[(size_t)prw * SEQ + pcl], 0, 1);
        }

        const half_t* krow = K + (size_t)(n0 + row) * HEAD_DIM;
        const v16h b0 = cat16(*(const v8h*)(krow + koff),      *(const v8h*)(krow + koff + 16));
        const v16h b1 = cat16(*(const v8h*)(krow + 32 + koff), *(const v8h*)(krow + 32 + koff + 16));
        v8f c = {};
        c = wmma_f16(a0, b0, c);
        c = wmma_f16(a1, b1, c);

        const int key = n0 + (lane & 15);
#pragma unroll
        for (int r = 0; r < 8; ++r) {
            const int qr = r + ((lane >> 4) << 3);    // local 0..15
            const int qg = q0 + qr;
            float lg = c[r] * scale + bias[((size_t)h * SEQ + qg) * SEQ + key];
            if (mask[(size_t)qg * SEQ + key] == 0) lg = -1e9f;
            logits[qr * LSTRIDE + key] = lg;
        }
    }
    __syncthreads();

    // ---- phase 2: softmax, 8 threads per row; interleaved columns c = part + 8*j ----
    // (8-lane row-groups touch 8 consecutive dwords -> conflict-light LDS, 32B-coalesced
    //  global writes of the attention-weight tensor)
    const int srow = tid >> 3, part = tid & 7;
    float* lrow = logits + srow * LSTRIDE;

    float mx = -3.4e38f;
    for (int j = 0; j < SEQ / 8; ++j) mx = fmaxf(mx, lrow[part + 8 * j]);
    scratch[tid] = mx;
    __syncthreads();
    float rmx = -3.4e38f;
#pragma unroll
    for (int p = 0; p < 8; ++p) rmx = fmaxf(rmx, scratch[(srow << 3) + p]);

    float sum = 0.f;
    for (int j = 0; j < SEQ / 8; ++j) sum += __expf(lrow[part + 8 * j] - rmx);
    scratch[128 + tid] = sum;
    __syncthreads();
    float rsum = 0.f;
#pragma unroll
    for (int p = 0; p < 8; ++p) rsum += scratch[128 + (srow << 3) + p];
    const float inv = 1.f / rsum;

    float* arow = attnw + ((size_t)bh * SEQ + q0 + srow) * SEQ;
    for (int j = 0; j < SEQ / 8; ++j) {
        const int c = part + 8 * j;
        const float p = __expf(lrow[c] - rmx) * inv;
        lrow[c] = p;
        arow[c] = p;
    }
    __syncthreads();

    // ---- phase 3: out = P @ V; each wave does a 16x16 chunk (cols wave*16..+15 of d) ----
    const int n0w = wave * 16;
    const half_t* vtrow = VT + (size_t)(n0w + (lane & 15)) * SEQ;   // B: lane = d-column, K = keys
    const float*  prow  = logits + (size_t)(lane & 15) * LSTRIDE;   // A: lane = query row

    v8f c = {};
    for (int ks = 0; ks < SEQ; ks += 32) {
        const int kb = ks + koff;
        v16h a;
        const float* p0 = prow + kb;
#pragma unroll
        for (int i = 0; i < 8; ++i) { a[i] = (half_t)p0[i]; a[8 + i] = (half_t)p0[16 + i]; }
        const v16h b = cat16(*(const v8h*)(vtrow + kb), *(const v8h*)(vtrow + kb + 16));
        c = wmma_f16(a, b, c);
    }
#pragma unroll
    for (int r = 0; r < 8; ++r) {
        const int qg = q0 + r + ((lane >> 4) << 3);
        out[((size_t)bh * SEQ + qg) * HEAD_DIM + n0w + (lane & 15)] = c[r];
    }
}

// ---------------- host launch ----------------
extern "C" void kernel_launch(void* const* d_in, const int* in_sizes, int n_in,
                              void* d_out, int out_size, void* d_ws, size_t ws_size,
                              hipStream_t stream) {
    (void)in_sizes; (void)n_in; (void)out_size; (void)ws_size;
    const float* x    = (const float*)d_in[0];
    const float* bias = (const float*)d_in[1];
    const int*   mask = (const int*)d_in[2];
    const float* wq   = (const float*)d_in[3];
    const float* bq   = (const float*)d_in[4];
    const float* wk   = (const float*)d_in[5];
    const float* bk   = (const float*)d_in[6];
    const float* wv   = (const float*)d_in[7];
    const float* bv   = (const float*)d_in[8];

    // workspace layout (halves)
    char* ws = (char*)d_ws;
    const size_t SZ_X  = (size_t)NTOK * D_MODEL;          // 4M halves
    const size_t SZ_W  = (size_t)D_MODEL * D_MODEL;       // 1M halves
    const size_t SZ_QKV = (size_t)BH * SEQ * HEAD_DIM;    // 4M halves
    half_t* x_h  = (half_t*)(ws);
    half_t* wq_h = (half_t*)(ws + SZ_X * 2);
    half_t* wk_h = (half_t*)(ws + (SZ_X + SZ_W) * 2);
    half_t* wv_h = (half_t*)(ws + (SZ_X + 2 * SZ_W) * 2);
    half_t* q_h  = (half_t*)(ws + (SZ_X + 3 * SZ_W) * 2);
    half_t* k_h  = (half_t*)(ws + (SZ_X + 3 * SZ_W + SZ_QKV) * 2);
    half_t* vt_h = (half_t*)(ws + (SZ_X + 3 * SZ_W + 2 * SZ_QKV) * 2);

    // d_out: [output 2*16*2048*64] then [atten 2*16*2048*2048], both fp32
    float* out   = (float*)d_out;
    float* attnw = out + (size_t)BH * SEQ * HEAD_DIM;

    // stage 1: converts
    {
        int n = (int)SZ_X;
        f32_to_f16_kernel<<<(n + 255) / 256, 256, 0, stream>>>(x, x_h, n);
        n = (int)SZ_W;
        f32_to_f16_kernel<<<(n + 255) / 256, 256, 0, stream>>>(wq, wq_h, n);
        f32_to_f16_kernel<<<(n + 255) / 256, 256, 0, stream>>>(wk, wk_h, n);
        f32_to_f16_kernel<<<(n + 255) / 256, 256, 0, stream>>>(wv, wv_h, n);
    }

    // stage 2: QKV projections (16384 tiles per matrix, 8 waves/block)
    qkv_gemm_kernel<<<dim3(2048, 3), 256, 0, stream>>>(
        x_h, wq_h, wk_h, wv_h, bq, bk, bv, q_h, k_h, vt_h);

    // stage 3: attention; padded logits + scratch dynamic LDS
    const size_t shmem = (size_t)(16 * LSTRIDE + 256) * sizeof(float);
    attn_kernel<<<dim3(SEQ / 16, BH), 128, shmem, stream>>>(
        q_h, k_h, vt_h, bias, mask, out, attnw);
}
